// GraphGIN_88072599372184
// MI455X (gfx1250) — matrix operand order repaired and checked by reading, not compiled, not run on hardware
//
#include <hip/hip_runtime.h>
#include <hip/hip_bf16.h>

typedef __bf16 bf16;
typedef __attribute__((ext_vector_type(16))) __bf16 v16bf;
typedef __attribute__((ext_vector_type(8)))  float  v8f;

#define N_NODES 100000
#define F_IN    128
#define H_DIM   64
#define C_OUT   2
#define E_EDGES 1000000
#define G_GRAPHS 1000

// ---------------- elementwise helpers ----------------

__global__ void k_copy_f4(const float* __restrict__ in, float* __restrict__ out, int n4) {
    int i = blockIdx.x * blockDim.x + threadIdx.x;
    if (i < n4) {
        ((float4*)out)[i] = ((const float4*)in)[i];
    }
}

__global__ void k_zero_f32(float* __restrict__ p, int n) {
    int i = blockIdx.x * blockDim.x + threadIdx.x;
    if (i < n) p[i] = 0.0f;
}

// transpose-convert weights: W (K x ncols, f32) -> Wt (ncols x K, bf16)
__global__ void k_w_transpose_bf16(const float* __restrict__ W, bf16* __restrict__ Wt,
                                   int K, int ncols) {
    int gid = blockIdx.x * blockDim.x + threadIdx.x;
    if (gid >= K * ncols) return;
    int k = gid / ncols;
    int n = gid - k * ncols;
    Wt[(size_t)n * K + k] = (bf16)W[gid];
}

// ---------------- edge scatter: z[dst,:] += h[src,:] ----------------
// one thread per (edge, 4-float chunk); float4 gather + 4 hw float atomics
__global__ void k_edge_scatter(const float* __restrict__ h, const int* __restrict__ ei,
                               float* __restrict__ z, int nedges, int feat) {
    int chunks = feat >> 2;
    int gid = blockIdx.x * blockDim.x + threadIdx.x;
    if (gid >= nedges * chunks) return;
    int e = gid / chunks;
    int c = gid - e * chunks;
    int s = ei[e];
    int d = ei[nedges + e];
    const float4 v = *(const float4*)(h + (size_t)s * feat + c * 4);
    float* p = z + (size_t)d * feat + c * 4;
    __hip_atomic_fetch_add(p + 0, v.x, __ATOMIC_RELAXED, __HIP_MEMORY_SCOPE_AGENT);
    __hip_atomic_fetch_add(p + 1, v.y, __ATOMIC_RELAXED, __HIP_MEMORY_SCOPE_AGENT);
    __hip_atomic_fetch_add(p + 2, v.z, __ATOMIC_RELAXED, __HIP_MEMORY_SCOPE_AGENT);
    __hip_atomic_fetch_add(p + 3, v.w, __ATOMIC_RELAXED, __HIP_MEMORY_SCOPE_AGENT);
}

// ---------------- WMMA GEMM ----------------
// out(nrows x ncols, f32) = act( A(nrows x K, f32->bf16 inline) @ Wt^T + bias )
// Wt is (ncols x K) bf16 (pre-transposed weights: row n = output column n, contiguous in K).
// One wave32 per 16-row strip; NT 16-wide column tiles per wave (A fragment reused NT times).
//
// v_wmma_f32_16x16x32_bf16 fragment layouts as loaded here, lane l: half=l>>4, m/n=l&15:
//   A 16x32:  elems 0..7  -> K = kk + half*16 + i      (contiguous)
//             elems 8..15 -> K = kk + half*16 + 8 + i  (contiguous)
//   B 32x16:  elems 0..15 -> K = kk + half*16 + i, column n  (contiguous in Wt row n)
//   C/D v8f:  elem j -> M = 16*tm + half*8 + j, N = 16*tn + (l&15)
template<int NT>
__global__ void k_wmma_gemm(const float* __restrict__ A, const bf16* __restrict__ Wt,
                            const float* __restrict__ bias, float* __restrict__ out,
                            int nrows, int K, int ncols, int relu) {
    int wid = blockIdx.x * (blockDim.x >> 5) + (threadIdx.x >> 5);
    int tiles_m = (nrows + 15) >> 4;
    if (wid >= tiles_m) return;   // wave-uniform: EXEC stays all-ones at WMMA

    int lane = threadIdx.x & 31;
    int half = lane >> 4;
    int mloc = lane & 15;

    int row = wid * 16 + mloc;
    int rclamp = row < nrows ? row : (nrows - 1);   // clamp: garbage rows never stored

    v8f acc[NT];
    #pragma unroll
    for (int t = 0; t < NT; ++t) acc[t] = (v8f){};

    for (int kk = 0; kk < K; kk += 32) {
        // A fragment: two contiguous 8-float groups, converted f32->bf16 inline
        const float* pa = A + (size_t)rclamp * K + kk + half * 16;
        v16bf a;
        #pragma unroll
        for (int i = 0; i < 8; ++i) {
            a[i]     = (bf16)pa[i];
            a[i + 8] = (bf16)pa[8 + i];
        }
        #pragma unroll
        for (int t = 0; t < NT; ++t) {
            int n = t * 16 + mloc;
            int nc = n < ncols ? n : (ncols - 1);   // clamp: garbage cols never stored
            const bf16* pb = Wt + (size_t)nc * K + kk + half * 16;
            v16bf b;
            #pragma unroll
            for (int i = 0; i < 16; ++i) b[i] = pb[i];   // 32 contiguous bytes
            acc[t] = __builtin_amdgcn_wmma_f32_16x16x32_bf16(
                         false, a, false, b, (short)0, acc[t], false, false);
        }
    }

    #pragma unroll
    for (int t = 0; t < NT; ++t) {
        int n = t * 16 + mloc;
        if (n < ncols) {
            float bv = bias[n];
            #pragma unroll
            for (int j = 0; j < 8; ++j) {
                int m = wid * 16 + half * 8 + j;
                if (m < nrows) {
                    float v = acc[t][j] + bv;
                    if (relu) v = v > 0.0f ? v : 0.0f;
                    out[(size_t)m * ncols + n] = v;
                }
            }
        }
    }
}

// ---------------- pooling ----------------

__global__ void k_pool_feat(const float* __restrict__ h, const int* __restrict__ batch,
                            float* __restrict__ sums, int n, int feat) {
    int gid = blockIdx.x * blockDim.x + threadIdx.x;
    if (gid >= n * feat) return;
    int i = gid / feat;
    int f = gid - i * feat;
    int g = batch[i];
    __hip_atomic_fetch_add(sums + (size_t)g * feat + f, h[(size_t)i * feat + f],
                           __ATOMIC_RELAXED, __HIP_MEMORY_SCOPE_AGENT);
}

__global__ void k_pool_cnt(const int* __restrict__ batch, float* __restrict__ counts, int n) {
    int i = blockIdx.x * blockDim.x + threadIdx.x;
    if (i < n) {
        __hip_atomic_fetch_add(counts + batch[i], 1.0f,
                               __ATOMIC_RELAXED, __HIP_MEMORY_SCOPE_AGENT);
    }
}

__global__ void k_pool_div(float* __restrict__ sums, const float* __restrict__ counts,
                           int g, int feat) {
    int gid = blockIdx.x * blockDim.x + threadIdx.x;
    if (gid >= g * feat) return;
    float c = counts[gid / feat];
    c = c > 1.0f ? c : 1.0f;
    sums[gid] = sums[gid] / c;
}

// ---------------- launch ----------------

static inline int cdiv(long long a, long long b) { return (int)((a + b - 1) / b); }

extern "C" void kernel_launch(void* const* d_in, const int* in_sizes, int n_in,
                              void* d_out, int out_size, void* d_ws, size_t ws_size,
                              hipStream_t stream) {
    const float* x   = (const float*)d_in[0];
    const int*   ei  = (const int*)d_in[1];
    const int*   bat = (const int*)d_in[2];
    const float* W1a = (const float*)d_in[3];
    const float* b1a = (const float*)d_in[4];
    const float* W1b = (const float*)d_in[5];
    const float* b1b = (const float*)d_in[6];
    const float* W2a = (const float*)d_in[7];
    const float* b2a = (const float*)d_in[8];
    const float* W2b = (const float*)d_in[9];
    const float* b2b = (const float*)d_in[10];
    const float* Wf  = (const float*)d_in[11];
    const float* bf_ = (const float*)d_in[12];
    float* outp = (float*)d_out;

    // workspace layout (256B aligned slabs)
    char* base = (char*)d_ws;
    size_t off = 0;
    auto alloc = [&](size_t bytes) -> char* {
        char* p = base + off;
        off = (off + bytes + 255) & ~(size_t)255;
        return p;
    };
    float* z    = (float*)alloc((size_t)N_NODES * F_IN * 4);   // reused as z2 (N x 64)
    float* t    = (float*)alloc((size_t)N_NODES * H_DIM * 4);
    float* h    = (float*)alloc((size_t)N_NODES * H_DIM * 4);
    float* sums = (float*)alloc((size_t)G_GRAPHS * H_DIM * 4);
    float* cnts = (float*)alloc((size_t)G_GRAPHS * 4);
    bf16*  W1at = (bf16*) alloc((size_t)H_DIM * F_IN * 2);     // (ncols x K) transposed bf16
    bf16*  W1bt = (bf16*) alloc((size_t)H_DIM * H_DIM * 2);
    bf16*  W2at = (bf16*) alloc((size_t)H_DIM * H_DIM * 2);
    bf16*  W2bt = (bf16*) alloc((size_t)H_DIM * H_DIM * 2);
    bf16*  Wft  = (bf16*) alloc((size_t)C_OUT * H_DIM * 2);
    (void)ws_size; (void)n_in; (void)in_sizes; (void)out_size;

    const int BS  = 256;
    const int WPB = BS / 32;

    // transpose-convert weights (tiny)
    k_w_transpose_bf16<<<cdiv(F_IN * H_DIM, BS), BS, 0, stream>>>(W1a, W1at, F_IN, H_DIM);
    k_w_transpose_bf16<<<cdiv(H_DIM * H_DIM, BS), BS, 0, stream>>>(W1b, W1bt, H_DIM, H_DIM);
    k_w_transpose_bf16<<<cdiv(H_DIM * H_DIM, BS), BS, 0, stream>>>(W2a, W2at, H_DIM, H_DIM);
    k_w_transpose_bf16<<<cdiv(H_DIM * H_DIM, BS), BS, 0, stream>>>(W2b, W2bt, H_DIM, H_DIM);
    k_w_transpose_bf16<<<cdiv(H_DIM * C_OUT, BS), BS, 0, stream>>>(Wf, Wft, H_DIM, C_OUT);

    // ---- layer 1 ----
    k_copy_f4<<<cdiv((long long)N_NODES * F_IN / 4, BS), BS, 0, stream>>>(x, z, N_NODES * F_IN / 4);
    k_edge_scatter<<<cdiv((long long)E_EDGES * (F_IN / 4), BS), BS, 0, stream>>>(x, ei, z, E_EDGES, F_IN);
    k_wmma_gemm<H_DIM / 16><<<cdiv(cdiv(N_NODES, 16), WPB), BS, 0, stream>>>(
        z, W1at, b1a, t, N_NODES, F_IN, H_DIM, 1);
    k_wmma_gemm<H_DIM / 16><<<cdiv(cdiv(N_NODES, 16), WPB), BS, 0, stream>>>(
        t, W1bt, b1b, h, N_NODES, H_DIM, H_DIM, 1);

    // ---- layer 2 ---- (reuse z/t as 64-wide buffers)
    k_copy_f4<<<cdiv((long long)N_NODES * H_DIM / 4, BS), BS, 0, stream>>>(h, z, N_NODES * H_DIM / 4);
    k_edge_scatter<<<cdiv((long long)E_EDGES * (H_DIM / 4), BS), BS, 0, stream>>>(h, ei, z, E_EDGES, H_DIM);
    k_wmma_gemm<H_DIM / 16><<<cdiv(cdiv(N_NODES, 16), WPB), BS, 0, stream>>>(
        z, W2at, b2a, t, N_NODES, H_DIM, H_DIM, 1);
    k_wmma_gemm<H_DIM / 16><<<cdiv(cdiv(N_NODES, 16), WPB), BS, 0, stream>>>(
        t, W2bt, b2b, h, N_NODES, H_DIM, H_DIM, 1);

    // ---- global mean pool ----
    k_zero_f32<<<cdiv(G_GRAPHS * H_DIM, BS), BS, 0, stream>>>(sums, G_GRAPHS * H_DIM);
    k_zero_f32<<<cdiv(G_GRAPHS, BS), BS, 0, stream>>>(cnts, G_GRAPHS);
    k_pool_feat<<<cdiv((long long)N_NODES * H_DIM, BS), BS, 0, stream>>>(h, bat, sums, N_NODES, H_DIM);
    k_pool_cnt<<<cdiv(N_NODES, BS), BS, 0, stream>>>(bat, cnts, N_NODES);
    k_pool_div<<<cdiv(G_GRAPHS * H_DIM, BS), BS, 0, stream>>>(sums, cnts, G_GRAPHS, H_DIM);

    // ---- classifier: (G x 64) @ (64 x 2) + bias -> d_out ----
    k_wmma_gemm<1><<<cdiv(cdiv(G_GRAPHS, 16), WPB), BS, 0, stream>>>(
        sums, Wft, bf_, outp, G_GRAPHS, H_DIM, C_OUT, 0);
}
